// STAN_86741159510396
// MI455X (gfx1250) — matrix operand order repaired and checked by reading, not compiled, not run on hardware
//
#include <hip/hip_runtime.h>
#include <hip/hip_bf16.h>

// ---------------- problem constants ----------------
#define NB   64      // batch
#define MM   128     // max_len
#define LL   8192    // l_dim
#define DD   64      // d_model
#define SUc  100.0f
#define TUc  500.0f

// ---------------- workspace layout (bytes) ----------------
#define WS_SCAL    0                        // 8 floats
#define WS_ROWSUM  4096                     // 8192 floats  (32 KB)
#define WS_Q       (WS_ROWSUM + 32768)      // N*M*D bf16   (1 MB)
#define WS_K       (WS_Q  + NB*MM*DD*2)
#define WS_VT      (WS_K  + NB*MM*DD*2)
#define WS_G       (WS_VT + NB*MM*DD*2)
#define WS_WOUT    (WS_G  + NB*MM*DD*2)     // L*D bf16     (1 MB)

// ---------------- vector types ----------------
typedef __attribute__((ext_vector_type(16))) __bf16         v16bf;
typedef __attribute__((ext_vector_type(16))) unsigned short v16u;
typedef __attribute__((ext_vector_type(8)))  unsigned short v8u;
typedef __attribute__((ext_vector_type(8)))  float          v8f;

__device__ __forceinline__ unsigned short f2bf(float f) {
    unsigned u = __builtin_bit_cast(unsigned, f);
    u = (u + 0x7FFFu + ((u >> 16) & 1u)) >> 16;      // round-to-nearest-even
    return (unsigned short)u;
}

// async copy of one 16-byte chunk: global (saddr + voff) -> LDS[ldsoff]
__device__ __forceinline__ void async_b128(unsigned ldsoff, unsigned voff,
                                           unsigned long long saddr) {
    asm volatile("global_load_async_to_lds_b128 %0, %1, %2"
                 :
                 : "v"(ldsoff), "v"(voff), "s"(saddr)
                 : "memory");
}
__device__ __forceinline__ void wait_async0() {
    asm volatile("s_wait_asynccnt 0x0" ::: "memory");
}

// A fragment (16x32 bf16): row = lane%16, K = 16*h + e  (16 contiguous)
__device__ __forceinline__ v16bf load_afrag(const unsigned short* base, int row,
                                            int stride, int kk, int h) {
    const unsigned short* r0 = base + row * stride + kk + 16 * h;
    const unsigned short* r1 = base + row * stride + kk + 16 * h + 8;
    v8u lo = *(const v8u*)r0;
    v8u hi = *(const v8u*)r1;
    v16u u;
#pragma unroll
    for (int i = 0; i < 8; ++i) { u[i] = lo[i]; u[8 + i] = hi[i]; }
    return __builtin_bit_cast(v16bf, u);
}

// B fragment (32x16 bf16): col = lane%16, K = kk + 16*h + e  (16 contiguous)
__device__ __forceinline__ v16bf load_bfrag(const unsigned short* base, int col,
                                            int stride, int kk, int h) {
    v16u u = *(const v16u*)(base + col * stride + kk + 16 * h);
    return __builtin_bit_cast(v16bf, u);
}

__device__ __forceinline__ v8f wmma_bf16(v16bf a, v16bf b, v8f c) {
    return __builtin_amdgcn_wmma_f32_16x16x32_bf16(false, a, false, b,
                                                   (short)0, c, false, false);
}

// ---------------- kernel: scalar embedding-row sums ----------------
__global__ void k_scal(const float* esl, const float* esu,
                       const float* etl, const float* etu, float* scal) {
    int t = threadIdx.x;
    if (t < 8) {
        int which = t >> 1, row = t & 1;
        const float* src = (which == 0 ? esl : which == 1 ? esu
                                              : which == 2 ? etl : etu) + row * DD;
        float s = 0.f;
        for (int i = 0; i < DD; ++i) s += src[i];
        scal[t] = s;
    }
}

// ---------------- kernel: W_out -> bf16 ----------------
__global__ void k_wout(const float* w, unsigned short* wbf) {
    int i = blockIdx.x * 256 + threadIdx.x;
    wbf[i] = f2bf(w[i]);
}

// ---------------- kernel: mat2s row sums (256 MB stream) ----------------
__global__ void k_rowsum(const float* mat2s, float* rowsum) {
    __shared__ float red[256];
    int l = blockIdx.x, t = threadIdx.x;
    const float4* p = (const float4*)(mat2s + (size_t)l * LL);
    float s = 0.f;
    for (int i = t; i < LL / 4; i += 256) {
        __builtin_prefetch(p + i + 512, 0, 0);
        float4 v = p[i];
        s += v.x + v.y + v.z + v.w;
    }
    red[t] = s;
    __syncthreads();
    for (int o = 128; o > 0; o >>= 1) {
        if (t < o) red[t] += red[t + o];
        __syncthreads();
    }
    if (t == 0) rowsum[l] = red[0];
}

// ---------------- kernel B1: joint + q/k/v projections ----------------
__global__ void k_b1(const int* traj, const float* emb_t, const float* emb_l,
                     const float* emb_u, const float* Wq, const float* Wk,
                     const float* Wv, unsigned short* qbf, unsigned short* kbf,
                     unsigned short* vtbf) {
    __shared__ __align__(64) unsigned short sj[MM * DD];      // joint bf16 16KB
    __shared__ __align__(64) unsigned short sw[3][DD * DD];   // Wq/Wk/Wv 24KB
    int n = blockIdx.x, tid = threadIdx.x;
    int lane = tid & 31, w = tid >> 5, h = lane >> 4, ln = lane & 15;

    const float* Ws[3] = {Wq, Wk, Wv};
    for (int idx = tid; idx < 3 * DD * DD; idx += 256) {
        int mi = idx / (DD * DD), r = idx % (DD * DD);
        sw[mi][r] = f2bf(Ws[mi][r]);
    }
    for (int idx = tid; idx < MM * DD; idx += 256) {
        int m = idx >> 6, d = idx & 63;
        int base = (n * MM + m) * 3;
        int user = traj[base + 0];
        int loc  = traj[base + 1];
        int tim  = traj[base + 2];
        int ti = (tim - 1) % 168 + 1;
        sj[idx] = f2bf(emb_t[ti * DD + d] + emb_l[loc * DD + d] + emb_u[user * DD + d]);
    }
    __syncthreads();

    v16bf af[2];
#pragma unroll
    for (int ks = 0; ks < 2; ++ks)
        af[ks] = load_afrag(sj, 16 * w + ln, DD, 32 * ks, h);

#pragma unroll
    for (int mi = 0; mi < 3; ++mi) {
#pragma unroll
        for (int t = 0; t < 4; ++t) {
            v8f acc = {};
#pragma unroll
            for (int ks = 0; ks < 2; ++ks) {
                v16bf bf = load_bfrag(sw[mi], 16 * t + ln, DD, 32 * ks, h);
                acc = wmma_bf16(af[ks], bf, acc);
            }
#pragma unroll
            for (int j = 0; j < 8; ++j) {
                int m = 16 * w + 8 * h + j;
                int d = 16 * t + ln;
                unsigned short v = f2bf(acc[j]);
                if (mi == 0)      qbf[n * MM * DD + m * DD + d] = v;
                else if (mi == 1) kbf[n * MM * DD + m * DD + d] = v;
                else              vtbf[n * MM * DD + d * MM + m] = v;  // transposed
            }
        }
    }
}

// ---------------- kernel B2: attention + fused deltas -> G ----------------
__global__ void k_b2(const int* traj, const int* traj_len, const float* mat1,
                     const float* vec, const float* emb_su, const float* emb_sl,
                     const float* emb_tu, const float* emb_tl, const float* scal,
                     const float* rowsum, const unsigned short* qbf,
                     const unsigned short* kbf, const unsigned short* vtbf,
                     unsigned short* gbf) {
    // [0,8192) q ; [8192,16384) k ; [16384,24576) vT ; attn overlays [0,16384)
    __shared__ __align__(64) unsigned short sm[3 * MM * DD];  // 48 KB
    int n = blockIdx.x, tid = threadIdx.x;
    int lane = tid & 31, w = tid >> 5, h = lane >> 4, ln = lane & 15;

    // async stage q / k / vT into LDS (no VGPR round-trip, ASYNCcnt tracked)
    {
        unsigned smbase = (unsigned)(size_t)(void*)sm;   // LDS byte offset
#pragma unroll
        for (int jsrc = 0; jsrc < 3; ++jsrc) {
            const unsigned short* src =
                (jsrc == 0 ? qbf : jsrc == 1 ? kbf : vtbf) + n * MM * DD;
            unsigned long long sa = (unsigned long long)(size_t)src;
            for (int i = tid; i < MM * DD / 8; i += 256) {   // 16B chunks
                async_b128(smbase + (unsigned)(jsrc * MM * DD * 2 + i * 16),
                           (unsigned)(i * 16), sa);
            }
        }
        wait_async0();
    }
    float ssl0 = scal[0], ssl1 = scal[1], ssu0 = scal[2], ssu1 = scal[3];
    float stl0 = scal[4], stl1 = scal[5], stu0 = scal[6], stu1 = scal[7];
    int tl = traj_len[n];
    __syncthreads();

    // ---- scores = q k^T + delta ----
    v16bf af[2];
#pragma unroll
    for (int ks = 0; ks < 2; ++ks)
        af[ks] = load_afrag(sm, 16 * w + ln, DD, 32 * ks, h);

    v8f sc[8];
#pragma unroll
    for (int t = 0; t < 8; ++t) {
        sc[t] = {};
#pragma unroll
        for (int ks = 0; ks < 2; ++ks) {
            v16bf bf = load_bfrag(sm + MM * DD, 16 * t + ln, DD, 32 * ks, h);
            sc[t] = wmma_bf16(af[ks], bf, sc[t]);
        }
#pragma unroll
        for (int j = 0; j < 8; ++j) {
            int m = 16 * w + 8 * h + j;
            int p = 16 * t + ln;
            float2 dd = *(const float2*)(mat1 + ((size_t)n * MM * MM + m * MM + p) * 2);
            bool msk = (m < tl) && (p < tl);
            float sslv = msk ? ssl1 : ssl0, ssuv = msk ? ssu1 : ssu0;
            float stlv = msk ? stl1 : stl0, stuv = msk ? stu1 : stu0;
            sc[t][j] += (sslv * (SUc - dd.x) + ssuv * dd.x) * (1.0f / SUc)
                      + (stlv * (TUc - dd.y) + stuv * dd.y) * (1.0f / TUc);
        }
    }

    // ---- softmax across 128 cols (rows live in this wave's registers) ----
    float rm[8], rs[8];
#pragma unroll
    for (int j = 0; j < 8; ++j) {
        float m = sc[0][j];
#pragma unroll
        for (int t = 1; t < 8; ++t) m = fmaxf(m, sc[t][j]);
#pragma unroll
        for (int o = 1; o < 16; o <<= 1) m = fmaxf(m, __shfl_xor(m, o, 16));
        rm[j] = m;
    }
#pragma unroll
    for (int t = 0; t < 8; ++t)
#pragma unroll
        for (int j = 0; j < 8; ++j) sc[t][j] = __expf(sc[t][j] - rm[j]);
#pragma unroll
    for (int j = 0; j < 8; ++j) {
        float s = 0.f;
#pragma unroll
        for (int t = 0; t < 8; ++t) s += sc[t][j];
#pragma unroll
        for (int o = 1; o < 16; o <<= 1) s += __shfl_xor(s, o, 16);
        rs[j] = 1.0f / s;
    }
    __syncthreads();  // everyone done reading q/k before attn overlays them

#pragma unroll
    for (int t = 0; t < 8; ++t)
#pragma unroll
        for (int j = 0; j < 8; ++j) {
            int m = 16 * w + 8 * h + j;
            int p = 16 * t + ln;
            float a = sc[t][j] * rs[j] * (((m < tl) && (p < tl)) ? 1.0f : 0.0f);
            sm[m * MM + p] = f2bf(a);
        }
    __syncthreads();

    // ---- self_attn = attn @ v, fused delta2 epilogue -> G (bf16) ----
    v16bf av[4];
#pragma unroll
    for (int ks = 0; ks < 4; ++ks)
        av[ks] = load_afrag(sm, 16 * w + ln, MM, 32 * ks, h);

#pragma unroll
    for (int t = 0; t < 4; ++t) {
        v8f oc = {};
#pragma unroll
        for (int ks = 0; ks < 4; ++ks) {
            v16bf bf = load_bfrag(sm + 2 * MM * DD, 16 * t + ln, MM, 32 * ks, h);
            oc = wmma_bf16(av[ks], bf, oc);
        }
#pragma unroll
        for (int j = 0; j < 8; ++j) {
            int m = 16 * w + 8 * h + j;
            int d = 16 * t + ln;
            int loc = traj[(n * MM + m) * 3 + 1];
            int m1 = (m < tl) ? 1 : 0;
            float dssum = (m < tl) ? rowsum[loc - 1] : 0.0f;
            float vv = vec[n * MM + m];
            float svsl = dssum;
            float svsu = SUc * (float)LL - dssum;
            float svtl = vv * (float)LL;
            float svtu = (TUc - vv) * (float)LL;
            float d2 = (emb_sl[m1 * DD + d] * svsu + emb_su[m1 * DD + d] * svsl) * (1.0f / SUc)
                     + (emb_tl[m1 * DD + d] * svtu + emb_tu[m1 * DD + d] * svtl) * (1.0f / TUc);
            gbf[(n * MM + m) * DD + d] = f2bf(oc[j] * d2);
        }
    }
}

// ---------------- kernel: out = G @ Wout^T + b  (8192x8192 fp32) ----------------
__global__ void k_gemm(const unsigned short* gbf, const unsigned short* wbf,
                       const float* b_out, float* out) {
    __shared__ __align__(64) unsigned short wt[128 * DD];  // 16 KB Wout tile
    int tid = threadIdx.x;
    int lane = tid & 31, w = tid >> 5, h = lane >> 4, ln = lane & 15;
    int r0 = blockIdx.y * 128, c0 = blockIdx.x * 128;

    // async stage the W_out tile into LDS
    {
        unsigned wtbase = (unsigned)(size_t)(void*)wt;
        unsigned long long sa = (unsigned long long)(size_t)(wbf + (size_t)c0 * DD);
        for (int i = tid; i < 128 * DD / 8; i += 256)
            async_b128(wtbase + (unsigned)(i * 16), (unsigned)(i * 16), sa);
        wait_async0();
    }
    __syncthreads();

    v16bf af[2];
#pragma unroll
    for (int ks = 0; ks < 2; ++ks)
        af[ks] = load_afrag(gbf, r0 + 16 * w + ln, DD, 32 * ks, h);

#pragma unroll
    for (int t = 0; t < 8; ++t) {
        v8f acc = {};
#pragma unroll
        for (int ks = 0; ks < 2; ++ks) {
            v16bf bf = load_bfrag(wt, 16 * t + ln, DD, 32 * ks, h);
            acc = wmma_bf16(af[ks], bf, acc);
        }
#pragma unroll
        for (int j = 0; j < 8; ++j) {
            int r = r0 + 16 * w + 8 * h + j;
            int c = c0 + 16 * t + ln;
            out[(size_t)r * LL + c] = acc[j] + b_out[c];
        }
    }
}

// ---------------- host launcher ----------------
extern "C" void kernel_launch(void* const* d_in, const int* in_sizes, int n_in,
                              void* d_out, int out_size, void* d_ws, size_t ws_size,
                              hipStream_t stream) {
    const int*   traj     = (const int*)d_in[0];
    const float* mat1     = (const float*)d_in[1];
    const float* mat2s    = (const float*)d_in[2];
    const float* vec      = (const float*)d_in[3];
    const int*   traj_len = (const int*)d_in[4];
    const float* emb_t    = (const float*)d_in[5];
    const float* emb_l    = (const float*)d_in[6];
    const float* emb_u    = (const float*)d_in[7];
    const float* emb_su   = (const float*)d_in[8];
    const float* emb_sl   = (const float*)d_in[9];
    const float* emb_tu   = (const float*)d_in[10];
    const float* emb_tl   = (const float*)d_in[11];
    const float* Wq       = (const float*)d_in[12];
    const float* Wk       = (const float*)d_in[13];
    const float* Wv       = (const float*)d_in[14];
    const float* W_out    = (const float*)d_in[15];
    const float* b_out    = (const float*)d_in[16];
    float* out = (float*)d_out;

    char* ws = (char*)d_ws;
    float*          scal   = (float*)(ws + WS_SCAL);
    float*          rowsum = (float*)(ws + WS_ROWSUM);
    unsigned short* qbf    = (unsigned short*)(ws + WS_Q);
    unsigned short* kbf    = (unsigned short*)(ws + WS_K);
    unsigned short* vtbf   = (unsigned short*)(ws + WS_VT);
    unsigned short* gbf    = (unsigned short*)(ws + WS_G);
    unsigned short* woutbf = (unsigned short*)(ws + WS_WOUT);

    k_scal<<<1, 32, 0, stream>>>(emb_sl, emb_su, emb_tl, emb_tu, scal);
    k_wout<<<(LL * DD) / 256, 256, 0, stream>>>(W_out, woutbf);
    k_rowsum<<<LL, 256, 0, stream>>>(mat2s, rowsum);
    k_b1<<<NB, 256, 0, stream>>>(traj, emb_t, emb_l, emb_u, Wq, Wk, Wv,
                                 qbf, kbf, vtbf);
    k_b2<<<NB, 256, 0, stream>>>(traj, traj_len, mat1, vec, emb_su, emb_sl,
                                 emb_tu, emb_tl, scal, rowsum, qbf, kbf, vtbf, gbf);
    k_gemm<<<dim3(LL / 128, (NB * MM) / 128), 256, 0, stream>>>(gbf, woutbf,
                                                                b_out, out);
}
